// OutputScalarNetwork_56873956933908
// MI455X (gfx1250) — compile-verified, matrix operand order checked by reading, not run on hardware
//
#include <hip/hip_runtime.h>
#include <hip/hip_bf16.h>
#include <math.h>

typedef __attribute__((ext_vector_type(16))) _Float16 v16h;
typedef __attribute__((ext_vector_type(8)))  float    v8f;

#define BB      16
#define NN      256
#define CC      64
#define RR      32
#define HH      128
#define NCHUNK  8
#define CHUNK   (NN / NCHUNK)          // 32 b's per wave
#define GAMMA_F 10.0f
#define SPACING (5.0f / 31.0f)         // CUTOFF/(N_RBF-1)
#define LOG2_F  0.69314718055994531f

// ---------------------------------------------------------------------------
// k0: zero the accumulator and per-batch bias-sum buffers in workspace
// ---------------------------------------------------------------------------
__global__ void k0_init(float* __restrict__ acc, float* __restrict__ S) {
    int idx = blockIdx.x * blockDim.x + threadIdx.x;
    if (idx < BB * NN) acc[idx] = 0.0f;
    if (idx < BB)      S[idx]   = 0.0f;
}

// ---------------------------------------------------------------------------
// k1: v[z,b,h] = sum_i W2[h,i] * feats[z,b,i] * mask[z,b] / 16
//     S[z]    += sum_i b2[i]   * feats[z,b,i] * mask[z,b] / 16   (once per b)
// ---------------------------------------------------------------------------
__global__ void k1_precompute_v(const float* __restrict__ feats,
                                const float* __restrict__ mask,
                                const float* __restrict__ W2,
                                const float* __restrict__ b2,
                                float* __restrict__ v,
                                float* __restrict__ S) {
    int idx = blockIdx.x * blockDim.x + threadIdx.x;   // over B*N*H
    if (idx >= BB * NN * HH) return;
    int h  = idx % HH;
    int zb = idx / HH;
    const float* f = feats + (size_t)zb * CC;
    const float* w = W2    + (size_t)h  * CC;
    float scale = mask[zb] * 0.0625f;                  // 1/sqrt(256)
    float s = 0.0f;
    #pragma unroll 8
    for (int i = 0; i < CC; ++i) s += w[i] * f[i];
    v[idx] = s * scale;
    if (h == 0) {
        float t = 0.0f;
        #pragma unroll 8
        for (int i = 0; i < CC; ++i) t += b2[i] * f[i];
        atomicAdd(&S[zb / NN], t * scale);
    }
}

// ---------------------------------------------------------------------------
// k2: main WMMA kernel.
// One wave (32 threads) handles (z, 16-row a-tile, 32-entry b-chunk).
// Per b: build A = rbf[16 a x 32 r] in f16, 8x WMMA vs W1 blocks [32r x 16h],
// relu, scale by v[z,b,hblk], accumulate; half-wave reduce; atomicAdd.
// ---------------------------------------------------------------------------
__global__ void __launch_bounds__(32)
k2_scfconv_wmma(const float* __restrict__ geom,
                const float* __restrict__ W1,
                const float* __restrict__ b1,
                const float* __restrict__ v,
                float* __restrict__ acc) {
    const int wid   = blockIdx.x;                    // B * (N/16) * NCHUNK
    const int chunk = wid % NCHUNK;
    const int aTile = (wid / NCHUNK) % (NN / 16);
    const int z     = wid / (NCHUNK * (NN / 16));

    const int lane = threadIdx.x;                    // wave32
    const int col  = lane & 15;                      // row M / col N within tile
    const int hi   = lane >> 4;                      // half-wave select
    const int a    = aTile * 16 + col;

    // K (rbf-center) indices for this lane's 8 packed f16 slots (ISA 16-bit
    // A 16x32 layout): slot t holds K = k0, k0+1 with
    //   k0 = 2t + (t>=4 ? 8 : 0) + 8*hi
    float centers[16];
    int   kidx[8];
    #pragma unroll
    for (int t = 0; t < 8; ++t) {
        int k0 = 2 * t + ((t >= 4) ? 8 : 0) + hi * 8;
        kidx[t] = k0;
        centers[2 * t]     = (float)k0       * SPACING;
        centers[2 * t + 1] = (float)(k0 + 1) * SPACING;
    }

    // Preload W1 as 8 B-fragments (B 32x16 f16 layout mirrors A: lane col = n,
    // same per-lane K slots). W1 is [32, 128] row-major.
    v16h w1f[8];
    #pragma unroll
    for (int blk = 0; blk < 8; ++blk) {
        const int n = blk * 16 + col;
        #pragma unroll
        for (int t = 0; t < 8; ++t) {
            const int k0 = kidx[t];
            w1f[blk][2 * t]     = (_Float16)W1[k0 * HH + n];
            w1f[blk][2 * t + 1] = (_Float16)W1[(k0 + 1) * HH + n];
        }
    }
    float b1v[8];
    #pragma unroll
    for (int blk = 0; blk < 8; ++blk) b1v[blk] = b1[blk * 16 + col];

    // This lane's atom a position
    const float* ga = geom + ((size_t)z * NN + a) * 3;
    const float ax = ga[0], ay = ga[1], az = ga[2];
    const float* gb_base   = geom + (size_t)z * NN * 3;
    const float* vrow_base = v + (size_t)z * NN * HH;

    float partial[8];
    #pragma unroll
    for (int j = 0; j < 8; ++j) partial[j] = 0.0f;

    const int b0 = chunk * CHUNK;
    for (int b = b0; b < b0 + CHUNK; ++b) {
        // prefetch next b's v-row into cache (global_prefetch_b8)
        if (b + 1 < b0 + CHUNK)
            __builtin_prefetch(vrow_base + (size_t)(b + 1) * HH + col, 0, 0);

        const float bx = gb_base[b * 3 + 0];
        const float by = gb_base[b * 3 + 1];
        const float bz = gb_base[b * 3 + 2];
        const float dx = ax - bx, dy = ay - by, dz = az - bz;
        const float r  = sqrtf(dx * dx + dy * dy + dz * dz + 1e-12f);

        // A fragment: 16 rbf values for row a (this lane), K per layout above
        v16h afrag;
        #pragma unroll
        for (int e = 0; e < 16; ++e) {
            const float d = r - centers[e];
            afrag[e] = (_Float16)__expf(-GAMMA_F * d * d);
        }

        const float* vrow = vrow_base + (size_t)b * HH;
        #pragma unroll
        for (int blk = 0; blk < 8; ++blk) {
            v8f c;
            #pragma unroll
            for (int j = 0; j < 8; ++j) c[j] = b1v[blk];
            v8f d = __builtin_amdgcn_wmma_f32_16x16x32_f16(
                        false, afrag, false, w1f[blk],
                        (short)0, c, false, false);
            const float vv = vrow[blk * 16 + col];
            #pragma unroll
            for (int j = 0; j < 8; ++j)
                partial[j] += fmaxf(d[j], 0.0f) * vv;
        }
    }

    // Row sums: D layout puts row M = j + 8*hi in VGPR j, columns across the
    // 16 lanes of each half-wave. XOR-reduce within each half (masks < 16
    // never cross the half boundary).
    #pragma unroll
    for (int j = 0; j < 8; ++j) {
        float s = partial[j];
        s += __shfl_xor(s, 1, 32);
        s += __shfl_xor(s, 2, 32);
        s += __shfl_xor(s, 4, 32);
        s += __shfl_xor(s, 8, 32);
        partial[j] = s;
    }
    float* accz = acc + (size_t)z * NN + aTile * 16 + hi * 8;
    #pragma unroll
    for (int j = 0; j < 8; ++j)
        if (col == j) atomicAdd(&accz[j], partial[j]);
}

// ---------------------------------------------------------------------------
// k3: out[z,a] = (softplus(acc + S[z]) - log2) * mask[z,a]
// ---------------------------------------------------------------------------
__global__ void k3_finalize(const float* __restrict__ acc,
                            const float* __restrict__ S,
                            const float* __restrict__ mask,
                            float* __restrict__ out) {
    int idx = blockIdx.x * blockDim.x + threadIdx.x;
    if (idx >= BB * NN) return;
    float x  = acc[idx] + S[idx / NN];
    float sp = fmaxf(x, 0.0f) + log1pf(__expf(-fabsf(x)));   // stable softplus
    out[idx] = (sp - LOG2_F) * mask[idx];
}

// ---------------------------------------------------------------------------
extern "C" void kernel_launch(void* const* d_in, const int* in_sizes, int n_in,
                              void* d_out, int out_size, void* d_ws, size_t ws_size,
                              hipStream_t stream) {
    const float* geom  = (const float*)d_in[0];   // [B,N,3]
    const float* feats = (const float*)d_in[1];   // [B,N,64]
    const float* mask  = (const float*)d_in[2];   // [B,N]
    const float* W1    = (const float*)d_in[3];   // [32,128]
    const float* b1    = (const float*)d_in[4];   // [128]
    const float* W2    = (const float*)d_in[5];   // [128,64]
    const float* b2    = (const float*)d_in[6];   // [64]
    float* out = (float*)d_out;                   // [B,N,1]

    float* v   = (float*)d_ws;                    // B*N*H floats (2 MB)
    float* acc = v + (size_t)BB * NN * HH;        // B*N floats
    float* S   = acc + BB * NN;                   // B floats

    k0_init<<<(BB * NN + 255) / 256, 256, 0, stream>>>(acc, S);
    k1_precompute_v<<<(BB * NN * HH + 127) / 128, 128, 0, stream>>>(
        feats, mask, W2, b2, v, S);
    k2_scfconv_wmma<<<BB * (NN / 16) * NCHUNK, 32, 0, stream>>>(
        geom, W1, b1, v, acc);
    k3_finalize<<<(BB * NN + 255) / 256, 256, 0, stream>>>(acc, S, mask, out);
}
